// LORI_FC_75419625718335
// MI455X (gfx1250) — matrix operand order repaired and checked by reading, not compiled, not run on hardware
//
#include <hip/hip_runtime.h>

// ---------------- types ----------------
typedef __bf16 bf16_t;
typedef __attribute__((ext_vector_type(16))) __bf16 bf16x16;
typedef __attribute__((ext_vector_type(8)))  float  f32x8;
typedef __attribute__((ext_vector_type(4)))  float  f32x4;

#define IN_F  4096
#define OUT_F 4096
#define RANK  64
#define BLK   64
#define NB    64
#define NROWS 8192  // 4 * 2048

#define WMMA_BF16(a, b, c) \
  __builtin_amdgcn_wmma_f32_16x16x32_bf16(false, (a), false, (b), (short)0, (c), false, false)

// ---------------- fragment loaders ----------------
// A-matrix 16x32 bf16 from f32 row-major memory.
// Lane layout (ISA 7.12.2): lanes 0-15 & 16-31 both hold M = lane&15.
// Element i: K = 16*(lane>>4) + i (contiguous, matches B fragment)
__device__ __forceinline__ bf16x16 frag_a_f32(const float* __restrict__ rowk, int khalf) {
  union { f32x4 v[4]; float f[16]; } u;
  const f32x4* p = (const f32x4*)(rowk + khalf * 16);
  const f32x4* q = (const f32x4*)(rowk + 8 + khalf * 16);
  u.v[0] = p[0]; u.v[1] = p[1];
  u.v[2] = q[0]; u.v[3] = q[1];
  bf16x16 r;
#pragma unroll
  for (int i = 0; i < 16; ++i) r[i] = (bf16_t)u.f[i];
  return r;
}

// Same, but non-temporal (last read of x: don't re-pollute L2).
__device__ __forceinline__ bf16x16 frag_a_f32_nt(const float* __restrict__ rowk, int khalf) {
  union { f32x4 v[4]; float f[16]; } u;
  const f32x4* p = (const f32x4*)(rowk + khalf * 16);
  const f32x4* q = (const f32x4*)(rowk + 8 + khalf * 16);
  u.v[0] = __builtin_nontemporal_load(p);
  u.v[1] = __builtin_nontemporal_load(p + 1);
  u.v[2] = __builtin_nontemporal_load(q);
  u.v[3] = __builtin_nontemporal_load(q + 1);
  bf16x16 r;
#pragma unroll
  for (int i = 0; i < 16; ++i) r[i] = (bf16_t)u.f[i];
  return r;
}

// B-matrix 32x16 bf16 from pre-converted bf16 stored as Bt[n][k] (k contiguous).
// Lane n: col = n&15, element i holds K = kbase + 16*(n>>4) + i -> one 32B load.
__device__ __forceinline__ bf16x16 frag_b_bf16(const bf16_t* __restrict__ bt, int kstride,
                                               int colbase, int kbase, int lane) {
  const int col = colbase + (lane & 15);
  const int kh  = lane >> 4;
  return *(const bf16x16*)(bt + (size_t)col * kstride + kbase + kh * 16);
}

// ---------------- kernel 0: weight prep ----------------
// bias_sum = bias + b_right; Wl, Wr -> bf16 (their [out][in] layout is already
// B-fragment friendly); diag[ob][k][n] -> diag_bf[ob][n][k] (transpose).
__global__ void lori_prep_kernel(const float* __restrict__ diag,
                                 const float* __restrict__ wl,
                                 const float* __restrict__ wr,
                                 const float* __restrict__ br,
                                 const float* __restrict__ bias,
                                 bf16_t* __restrict__ diag_bf,
                                 bf16_t* __restrict__ wl_bf,
                                 bf16_t* __restrict__ wr_bf,
                                 float*  __restrict__ bias_sum) {
  const int id = blockIdx.x * blockDim.x + threadIdx.x;
  const int stride = gridDim.x * blockDim.x;
  const int n = RANK * IN_F;  // 262144, same count for all three weight arrays
  for (int i = id; i < n; i += stride) {
    wl_bf[i] = (bf16_t)wl[i];
    wr_bf[i] = (bf16_t)wr[i];
    const int ob = i >> 12, c = (i >> 6) & 63, k = i & 63;
    diag_bf[i] = (bf16_t)diag[((ob * BLK) + k) * BLK + c];
  }
  for (int i = id; i < OUT_F; i += stride) bias_sum[i] = bias[i] + br[i];
}

// ---------------- kernel 1: t = x @ Wl^T  (M=8192, N=64, K=4096) ----------------
// 128 threads = 4 waves per block; block owns 32 rows; wave w owns K chunk
// [w*1024, w*1024+1024). 32 rows/wave halves B-fragment traffic per WMMA.
// Software-pipelined: fragments for step s+1 are loaded before the WMMAs of
// step s, so s_wait_loadcnt lands a full WMMA block after the loads.
__global__ void __launch_bounds__(128) lori_rank_kernel(const float* __restrict__ x,
                                                        const bf16_t* __restrict__ wl_bf,
                                                        float* __restrict__ t) {
  __shared__ float red[4][32][RANK];  // 32 KB
  const int tid = threadIdx.x;
  const int lane = tid & 31;
  const int w = tid >> 5;
  const int rowbase = blockIdx.x * 32;
  const int mrow = lane & 15;
  const int khalf = lane >> 4;
  const float* rowptr0 = x + (size_t)(rowbase + mrow) * IN_F;
  const float* rowptr1 = rowptr0 + (size_t)16 * IN_F;
  const int kchunk = w * (IN_F / 4);

  f32x8 acc[2][4] = {};
  bf16x16 a0 = frag_a_f32(rowptr0 + kchunk, khalf);
  bf16x16 a1 = frag_a_f32(rowptr1 + kchunk, khalf);
  bf16x16 b[4];
#pragma unroll
  for (int cg = 0; cg < 4; ++cg) b[cg] = frag_b_bf16(wl_bf, IN_F, cg * 16, kchunk, lane);

#pragma unroll 2
  for (int s = 0; s < 31; ++s) {
    const int kn = kchunk + (s + 1) * 32;
    // stream-ahead prefetch of x (one 128B line per row, ~7 steps ahead);
    // overruns at the chunk edge are speculative and dropped per ISA.
    __builtin_prefetch(rowptr0 + kn + 224, 0, 3);
    __builtin_prefetch(rowptr1 + kn + 224, 0, 3);
    const bf16x16 na0 = frag_a_f32(rowptr0 + kn, khalf);
    const bf16x16 na1 = frag_a_f32(rowptr1 + kn, khalf);
    bf16x16 nb[4];
#pragma unroll
    for (int cg = 0; cg < 4; ++cg) nb[cg] = frag_b_bf16(wl_bf, IN_F, cg * 16, kn, lane);
#pragma unroll
    for (int cg = 0; cg < 4; ++cg) {
      acc[0][cg] = WMMA_BF16(a0, b[cg], acc[0][cg]);
      acc[1][cg] = WMMA_BF16(a1, b[cg], acc[1][cg]);
    }
    a0 = na0; a1 = na1;
#pragma unroll
    for (int cg = 0; cg < 4; ++cg) b[cg] = nb[cg];
  }
#pragma unroll
  for (int cg = 0; cg < 4; ++cg) {
    acc[0][cg] = WMMA_BF16(a0, b[cg], acc[0][cg]);
    acc[1][cg] = WMMA_BF16(a1, b[cg], acc[1][cg]);
  }

  const int colq = lane & 15;
  const int rh = (lane >> 4) * 8;
#pragma unroll
  for (int rg = 0; rg < 2; ++rg)
#pragma unroll
    for (int cg = 0; cg < 4; ++cg)
#pragma unroll
      for (int r = 0; r < 8; ++r)
        red[w][rg * 16 + rh + r][cg * 16 + colq] = acc[rg][cg][r];
  __syncthreads();
  const float* rp = &red[0][0][0];
  for (int e = tid; e < 32 * RANK; e += 128) {
    const float sum = rp[e] + rp[2048 + e] + rp[4096 + e] + rp[6144 + e];
    t[(size_t)(rowbase + (e >> 6)) * RANK + (e & 63)] = sum;
  }
}

// ---------------- kernel 2: out = blockdiag(x) + t @ Wr^T + bias_sum ----------------
// Grid (64 row tiles, 64 output blocks); 4 waves/block; wave = 32 rows x 64 cols,
// so each B1/B2 fragment feeds 2 WMMAs.
__global__ void __launch_bounds__(128) lori_out_kernel(const float* __restrict__ x,
                                                       const float* __restrict__ t,
                                                       const bf16_t* __restrict__ diag_bf,
                                                       const bf16_t* __restrict__ wr_bf,
                                                       const float* __restrict__ bias_sum,
                                                       float* __restrict__ out) {
  const int tid = threadIdx.x;
  const int lane = tid & 31;
  const int w = tid >> 5;
  const int ob = blockIdx.y;
  const int rowbase = blockIdx.x * 128 + w * 32;
  const int mrow = lane & 15;
  const int khalf = lane >> 4;

  const float* xrow0 = x + (size_t)(rowbase + mrow) * IN_F + ob * BLK;  // diag slice only
  const float* xrow1 = xrow0 + (size_t)16 * IN_F;
  const float* trow0 = t + (size_t)(rowbase + mrow) * RANK;
  const float* trow1 = trow0 + (size_t)16 * RANK;
  const bf16_t* dblk = diag_bf + (size_t)ob * BLK * BLK;        // [n][k]
  const bf16_t* wrb  = wr_bf   + (size_t)ob * BLK * RANK;       // [n][k]

  f32x8 acc[2][4] = {};
#pragma unroll
  for (int kb = 0; kb < 64; kb += 32) {
    const bf16x16 xa0 = frag_a_f32_nt(xrow0 + kb, khalf);  // last use of x
    const bf16x16 xa1 = frag_a_f32_nt(xrow1 + kb, khalf);
    const bf16x16 ta0 = frag_a_f32(trow0 + kb, khalf);     // t is L2-hot
    const bf16x16 ta1 = frag_a_f32(trow1 + kb, khalf);
#pragma unroll
    for (int cg = 0; cg < 4; ++cg) {
      const bf16x16 b1 = frag_b_bf16(dblk, BLK, cg * 16, kb, lane);
      const bf16x16 b2 = frag_b_bf16(wrb, RANK, cg * 16, kb, lane);
      acc[0][cg] = WMMA_BF16(xa0, b1, acc[0][cg]);
      acc[0][cg] = WMMA_BF16(ta0, b2, acc[0][cg]);
      acc[1][cg] = WMMA_BF16(xa1, b1, acc[1][cg]);
      acc[1][cg] = WMMA_BF16(ta1, b2, acc[1][cg]);
    }
  }

  const int colq = lane & 15;
  const int rh = (lane >> 4) * 8;
#pragma unroll
  for (int rg = 0; rg < 2; ++rg)
#pragma unroll
    for (int cg = 0; cg < 4; ++cg) {
      const int col = ob * BLK + cg * 16 + colq;
      const float bs = bias_sum[col];
#pragma unroll
      for (int r = 0; r < 8; ++r) {
        const float v = acc[rg][cg][r] + bs;
        // streaming store: keep x/t resident in L2 instead of the output
        __builtin_nontemporal_store(v, out + (size_t)(rowbase + rg * 16 + rh + r) * OUT_F + col);
      }
    }
}

// ---------------- launch ----------------
extern "C" void kernel_launch(void* const* d_in, const int* in_sizes, int n_in,
                              void* d_out, int out_size, void* d_ws, size_t ws_size,
                              hipStream_t stream) {
  (void)in_sizes; (void)n_in; (void)out_size; (void)ws_size;
  const float* x    = (const float*)d_in[0];
  const float* diag = (const float*)d_in[1];
  const float* wl   = (const float*)d_in[2];
  const float* wr   = (const float*)d_in[3];
  const float* br   = (const float*)d_in[4];
  const float* bias = (const float*)d_in[5];
  float* out = (float*)d_out;

  char* ws = (char*)d_ws;
  float*  t       = (float*)(ws);                                  // 2 MB
  bf16_t* wl_bf   = (bf16_t*)(ws + (size_t)(2 << 20));             // 512 KB
  bf16_t* wr_bf   = (bf16_t*)(ws + (size_t)(2 << 20) + (512 << 10));
  bf16_t* diag_bf = (bf16_t*)(ws + (size_t)(2 << 20) + (1024 << 10));
  float*  bsum    = (float*)(ws + (size_t)(2 << 20) + (1536 << 10));

  lori_prep_kernel<<<256, 256, 0, stream>>>(diag, wl, wr, br, bias,
                                            diag_bf, wl_bf, wr_bf, bsum);
  lori_rank_kernel<<<NROWS / 32, 128, 0, stream>>>(x, wl_bf, t);
  dim3 gB(NROWS / 128, NB);
  lori_out_kernel<<<gB, 128, 0, stream>>>(x, t, diag_bf, wr_bf, bsum, out);
}
